// Rerank_model_44418551775905
// MI455X (gfx1250) — matrix operand order repaired and checked by reading, not compiled, not run on hardware
//
#include <hip/hip_runtime.h>

// ---------------------------------------------------------------------------
// Problem constants (match reference)
// ---------------------------------------------------------------------------
#define NN      50000
#define EE      800000
#define BB      64
#define DIN_X   29
#define DH      128
#define DIN_E   17
#define DE      64
#define ZW      (2 * DH + DE)   // 320
#define KSTEPS  (ZW / 32)       // 10

typedef float v8f  __attribute__((ext_vector_type(8)));
typedef __bf16 v16bf __attribute__((ext_vector_type(16)));

// f32 -> bf16 (round to nearest even), stored as ushort
static __device__ __forceinline__ unsigned short f32_to_bf16(float f) {
    unsigned int u = __float_as_uint(f);
    u += 0x7FFFu + ((u >> 16) & 1u);
    return (unsigned short)(u >> 16);
}

static __device__ __forceinline__ float eluf(float x) {
    return x > 0.0f ? x : expm1f(x);
}

// byte offset of a generic pointer known to live in LDS (as(3) ptr value ==
// hardware DS offset used by ds_* / async-to-LDS instructions)
static __device__ __forceinline__ unsigned lds_off(const void* p) {
    return (unsigned)(size_t)(const __attribute__((address_space(3))) void*)p;
}

// async 16-byte gather: global -> LDS, tracked with ASYNCcnt
static __device__ __forceinline__ void async_copy_b128(unsigned ldsOff,
                                                       const void* gaddr) {
    asm volatile("global_load_async_to_lds_b128 %0, %1, off"
                 :: "v"(ldsOff), "v"((unsigned long long)(size_t)gaddr)
                 : "memory");
}

static __device__ __forceinline__ void wait_asynccnt0() {
    asm volatile("s_wait_asynccnt 0x0" ::: "memory");
}

// ---------------------------------------------------------------------------
// Zero a float buffer
// ---------------------------------------------------------------------------
__global__ void k_zero(float* __restrict__ p, long n) {
    long i = (long)blockIdx.x * blockDim.x + threadIdx.x;
    long stride = (long)gridDim.x * blockDim.x;
    for (; i < n; i += stride) p[i] = 0.0f;
}

// ---------------------------------------------------------------------------
// Node embedding: h = elu(x @ Wx + bx) -> h f32 [N,128] and hbf bf16 [N,128]
// ---------------------------------------------------------------------------
__global__ void k_embed_nodes(const float* __restrict__ x,
                              const float* __restrict__ Wx,
                              const float* __restrict__ bx,
                              float* __restrict__ h,
                              unsigned short* __restrict__ hbf) {
    long idx = (long)blockIdx.x * blockDim.x + threadIdx.x;
    long total = (long)NN * DH;
    if (idx >= total) return;
    int n = (int)(idx / DH);
    int c = (int)(idx % DH);
    const float* xr = x + (long)n * DIN_X;
    float acc = bx[c];
#pragma unroll
    for (int k = 0; k < DIN_X; ++k) acc += xr[k] * Wx[k * DH + c];
    acc = eluf(acc);
    h[idx] = acc;
    hbf[idx] = f32_to_bf16(acc);
}

// ---------------------------------------------------------------------------
// Edge message kernel (the 262-GFLOP hot loop).
// Persistent 256-thread workgroups (8 wave32).
//  - Per-wave B (weight) fragments for both gate matrices held in VGPRs
//    (v16bf regF/regS[10] ~ 160 VGPRs), loaded once per wave from the
//    L2-resident f32 weights -> hot loop touches LDS only for A fragments.
//  - Double-buffered 16-edge z tiles: h[dst]/h[src] gathered with
//    GLOBAL_LOAD_ASYNC_TO_LDS_B128 (ASYNCcnt) issued one tile ahead so the
//    gather latency overlaps the WMMA loop; edge-embed slice computed in
//    VALU and packed-stored. One barrier per tile. LDS ~25 KB -> multiple
//    workgroups per WGP.
//  - 8 waves x 2 matrices x (K=320 -> 10 steps) v_wmma_f32_16x16x32_bf16.
//  - Epilogue: ew * sigmoid(f) * softplus(s), scattered with global f32
//    atomics (segment_sum by dst).
// ---------------------------------------------------------------------------
__global__ __launch_bounds__(256)
void k_msg(const unsigned short* __restrict__ hbf,
           const float* __restrict__ edge_attr,
           const int*  __restrict__ edge_index,     // [2,E]: src row then dst row
           const float* __restrict__ We,            // [17,64]
           const float* __restrict__ bee,           // [64]
           const float* __restrict__ Wf,            // [320,128]
           const float* __restrict__ bf,            // [128]
           const float* __restrict__ Ws,            // [320,128]
           const float* __restrict__ bs,            // [128]
           float* __restrict__ agg,                 // [N,128] f32, pre-zeroed
           int mode) {
    extern __shared__ unsigned char smemRaw[];
    unsigned short* sZ = (unsigned short*)smemRaw;         // 2 x [16][320] bf16
    float* sWe  = (float*)(sZ + 2 * 16 * ZW);              // [17*64]
    float* sBee = sWe + DIN_E * DE;                        // [64]
    float* sEw  = sBee + DE;                               // 2 x [16]
    int*   sDst = (int*)(sEw + 32);                        // 2 x [16]

    const int tid  = threadIdx.x;
    const int lane = tid & 31;
    const int wid  = tid >> 5;          // 0..7 : N-tile id
    const int laneHalf = lane >> 4;     // 0/1
    const int lcol = lane & 15;
    const int col  = wid * 16 + lcol;   // output channel of this lane

    // ---- small LDS preload: edge-embed weights
    for (int idx = tid; idx < DIN_E * DE; idx += 256) sWe[idx] = We[idx];
    if (tid < DE) sBee[tid] = bee[tid];

    // ---- hoist B fragments (both gate matrices) into VGPRs, once per wave.
    // B (32x16 bf16) layout: lane<16 holds K=kb..kb+15 of its column,
    // lane>=16 holds K=kb+16..kb+31. Weights are row-major [k][n] f32.
    v16bf regF[KSTEPS];
    v16bf regS[KSTEPS];
#pragma unroll
    for (int kk = 0; kk < KSTEPS; ++kk) {
        const int kb = kk * 32 + laneHalf * 16;
        unsigned* pf = (unsigned*)&regF[kk];
        unsigned* ps = (unsigned*)&regS[kk];
#pragma unroll
        for (int q = 0; q < 8; ++q) {
            const int k0 = kb + 2 * q;
            unsigned f0 = f32_to_bf16(Wf[(size_t)k0 * DH + col]);
            unsigned f1 = f32_to_bf16(Wf[(size_t)(k0 + 1) * DH + col]);
            pf[q] = f0 | (f1 << 16);
            unsigned s0 = f32_to_bf16(Ws[(size_t)k0 * DH + col]);
            unsigned s1 = f32_to_bf16(Ws[(size_t)(k0 + 1) * DH + col]);
            ps[q] = s0 | (s1 << 16);
        }
    }
    const float bfv = bf[col];
    const float bsv = bs[col];
    __syncthreads();

    const int* srcIdx = edge_index;
    const int* dstIdx = edge_index + EE;
    const int ntiles = EE / 16;
    const int G = gridDim.x;

    // stage tile t into z-buffer `buf` (async gathers + VALU edge-embed)
    auto issue = [&](int t, int buf) {
        const int r = tid >> 4;          // edge row in tile
        const int j = tid & 15;          // worker within row
        const long e = (long)t * 16 + r;
        const int sN = srcIdx[e];
        const int dN = dstIdx[e];
        unsigned short* z = sZ + buf * 16 * ZW + r * ZW;
        const unsigned zoff = lds_off(z);
        // z[0:128) = h[dst], z[128:256) = h[src]: async b128 per worker
        async_copy_b128(zoff + j * 16,       hbf + (size_t)dN * DH + j * 8);
        async_copy_b128(zoff + 256 + j * 16, hbf + (size_t)sN * DH + j * 8);
        // z[256:320) = elu(edge_attr @ We + bee): 4 channels, packed store
        const float* ea = edge_attr + e * DIN_E;
        const float ea0 = ea[0];
        float c[4];
#pragma unroll
        for (int cc = 0; cc < 4; ++cc) {
            int ch = j * 4 + cc;
            float acc = sBee[ch];
#pragma unroll
            for (int k = 0; k < DIN_E; ++k) acc += ea[k] * sWe[k * DE + ch];
            c[cc] = eluf(acc);
        }
        uint2 pk;
        pk.x = (unsigned)f32_to_bf16(c[0]) | ((unsigned)f32_to_bf16(c[1]) << 16);
        pk.y = (unsigned)f32_to_bf16(c[2]) | ((unsigned)f32_to_bf16(c[3]) << 16);
        *(uint2*)&z[2 * DH + j * 4] = pk;
        if (j == 0) {
            sDst[buf * 16 + r] = dN;
            float cov = (ea0 > 0.5f) ? 1.0f : 0.0f;
            sEw[buf * 16 + r] = mode ? (1.0f - cov) : cov;
        }
    };

    int t = blockIdx.x;
    int buf = 0;
    if (t < ntiles) issue(t, 0);

    for (; t < ntiles; t += G, buf ^= 1) {
        // own-wave async gathers done; barrier makes all waves' writes visible
        wait_asynccnt0();
        __syncthreads();

        // prefetch next tile into the other buffer (overlaps WMMA below)
        const int tn = t + G;
        if (tn < ntiles) issue(tn, buf ^ 1);

        // ---------- WMMA: D[16 edges x 16 outs] over K=320 for f and s ------
        v8f accF = {};
        v8f accS = {};
        const unsigned short* zr = sZ + buf * 16 * ZW + lcol * ZW;  // A row
#pragma unroll
        for (int kk = 0; kk < KSTEPS; ++kk) {
            const int kb = kk * 32;
            v16bf a;
            // A (16x32 bf16): lane<16 K=kb..kb+15; lane>=16 K=kb+16..kb+31
            ((uint4*)&a)[0] = *(const uint4*)(zr + kb + laneHalf * 16);
            ((uint4*)&a)[1] = *(const uint4*)(zr + kb + laneHalf * 16 + 8);
            accF = __builtin_amdgcn_wmma_f32_16x16x32_bf16(
                false, a, false, regF[kk], (short)0, accF, false, false);
            accS = __builtin_amdgcn_wmma_f32_16x16x32_bf16(
                false, a, false, regS[kk], (short)0, accS, false, false);
        }

        // ---------- epilogue: gate, weight, scatter-add (segment_sum) -------
#pragma unroll
        for (int i = 0; i < 8; ++i) {
            const int rr = i + laneHalf * 8;   // edge row per C/D layout
            float fg = accF[i] + bfv;
            float sg = accS[i] + bsv;
            float sig = 1.0f / (1.0f + expf(-fg));
            float sp  = (sg > 20.0f) ? sg : log1pf(expf(sg));
            float m = sEw[buf * 16 + rr] * sig * sp;
            atomicAdd(&agg[(size_t)sDst[buf * 16 + rr] * DH + col], m);
        }
    }
}

// ---------------------------------------------------------------------------
// Post block: h_out = elu( LN(agg + h_in) @ Wl + bl + h_in )
// one 128-thread workgroup per node
// ---------------------------------------------------------------------------
__global__ __launch_bounds__(128)
void k_post(const float* __restrict__ agg,
            const float* __restrict__ hin,
            const float* __restrict__ g,
            const float* __restrict__ beta,
            const float* __restrict__ Wl,
            const float* __restrict__ bl,
            float* __restrict__ hout,
            unsigned short* __restrict__ hbf_out) {
    __shared__ float sLN[DH];
    __shared__ float sRed[DH];
    const int i = blockIdx.x;
    const int t = threadIdx.x;
    const float hv = hin[(size_t)i * DH + t];
    const float v  = agg[(size_t)i * DH + t] + hv;

    sRed[t] = v;
    __syncthreads();
    for (int off = 64; off > 0; off >>= 1) {
        if (t < off) sRed[t] += sRed[t + off];
        __syncthreads();
    }
    const float mean = sRed[0] * (1.0f / DH);
    __syncthreads();
    const float dv = v - mean;
    sRed[t] = dv * dv;
    __syncthreads();
    for (int off = 64; off > 0; off >>= 1) {
        if (t < off) sRed[t] += sRed[t + off];
        __syncthreads();
    }
    const float var = sRed[0] * (1.0f / DH);
    const float ln  = dv * rsqrtf(var + 1e-5f) * g[t] + beta[t];
    sLN[t] = ln;
    __syncthreads();

    float acc = bl[t];
#pragma unroll 8
    for (int k = 0; k < DH; ++k) acc += sLN[k] * Wl[k * DH + t];
    acc += hv;
    const float out = eluf(acc);
    hout[(size_t)i * DH + t] = out;
    hbf_out[(size_t)i * DH + t] = f32_to_bf16(out);
}

// ---------------------------------------------------------------------------
// Readout MLP + masked pooled segment-sum. one 64-thread workgroup per node.
// ---------------------------------------------------------------------------
__global__ __launch_bounds__(64)
void k_readout(const float* __restrict__ h,
               const float* __restrict__ x,
               const int*  __restrict__ batch,
               const float* __restrict__ Wm1, const float* __restrict__ bm1,
               const float* __restrict__ Wm2, const float* __restrict__ bm2,
               float* __restrict__ pooled) {
    __shared__ float sh[DH];
    __shared__ float sm1[DH / 2];
    const int i = blockIdx.x;
    const int t = threadIdx.x;
    sh[t]      = h[(size_t)i * DH + t];
    sh[t + 64] = h[(size_t)i * DH + t + 64];
    __syncthreads();
    float a = bm1[t];
#pragma unroll 8
    for (int k = 0; k < DH; ++k) a += sh[k] * Wm1[k * (DH / 2) + t];
    sm1[t] = eluf(a);
    __syncthreads();
    if (t < DH / 4) {
        float b = bm2[t];
#pragma unroll 8
        for (int k = 0; k < DH / 2; ++k) b += sm1[k] * Wm2[k * (DH / 4) + t];
        const float m2 = eluf(b);
        if (x[(size_t)i * DIN_X] > 0.5f) {          // lig_mask
            atomicAdd(&pooled[(size_t)batch[i] * (DH / 4) + t], m2);
        }
    }
}

// ---------------------------------------------------------------------------
// energy[b] = (pooled[b] . Wt) / n_atom[b]
// ---------------------------------------------------------------------------
__global__ void k_energy(const float* __restrict__ pooled,
                         const float* __restrict__ Wt,
                         const float* __restrict__ n_atom,
                         float* __restrict__ out) {
    const int b = threadIdx.x;
    if (b >= BB) return;
    float e = 0.0f;
#pragma unroll
    for (int c = 0; c < DH / 4; ++c) e += pooled[b * (DH / 4) + c] * Wt[c];
    out[b] = e / n_atom[b];
}

// ---------------------------------------------------------------------------
// Host-side launcher
// ---------------------------------------------------------------------------
extern "C" void kernel_launch(void* const* d_in, const int* in_sizes, int n_in,
                              void* d_out, int out_size, void* d_ws, size_t ws_size,
                              hipStream_t stream) {
    (void)in_sizes; (void)n_in; (void)out_size; (void)ws_size;

    const float* x         = (const float*)d_in[0];
    const float* edge_attr = (const float*)d_in[1];
    const int*   edge_idx  = (const int*)  d_in[2];
    const int*   batch     = (const int*)  d_in[3];
    const float* n_atom    = (const float*)d_in[4];
    const float* Wx  = (const float*)d_in[5];
    const float* bx  = (const float*)d_in[6];
    const float* We  = (const float*)d_in[7];
    const float* bee = (const float*)d_in[8];
    const float* Wf1 = (const float*)d_in[9];
    const float* bf1 = (const float*)d_in[10];
    const float* Ws1 = (const float*)d_in[11];
    const float* bs1 = (const float*)d_in[12];
    const float* g1    = (const float*)d_in[13];
    const float* beta1 = (const float*)d_in[14];
    const float* Wl1 = (const float*)d_in[15];
    const float* bl1 = (const float*)d_in[16];
    const float* Wf2 = (const float*)d_in[17];
    const float* bf2 = (const float*)d_in[18];
    const float* Ws2 = (const float*)d_in[19];
    const float* bs2 = (const float*)d_in[20];
    const float* g2    = (const float*)d_in[21];
    const float* beta2 = (const float*)d_in[22];
    const float* Wl2 = (const float*)d_in[23];
    const float* bl2 = (const float*)d_in[24];
    const float* Wm1 = (const float*)d_in[25];
    const float* bm1 = (const float*)d_in[26];
    const float* Wm2 = (const float*)d_in[27];
    const float* bm2 = (const float*)d_in[28];
    const float* Wt  = (const float*)d_in[29];

    float* out = (float*)d_out;

    // ---- workspace layout (256-byte aligned regions) ----
    unsigned char* ws = (unsigned char*)d_ws;
    unsigned short* hbf = (unsigned short*)ws;                    // N*128 bf16
    size_t off = (size_t)NN * DH * sizeof(unsigned short);        // 12.8 MB
    off = (off + 255) & ~(size_t)255;
    float* agg = (float*)(ws + off); off += (size_t)NN * DH * sizeof(float);
    off = (off + 255) & ~(size_t)255;
    float* hA  = (float*)(ws + off); off += (size_t)NN * DH * sizeof(float);
    off = (off + 255) & ~(size_t)255;
    float* hB  = (float*)(ws + off); off += (size_t)NN * DH * sizeof(float);
    off = (off + 255) & ~(size_t)255;
    float* pooled = (float*)(ws + off);                           // B * 32

    const long nodeElems = (long)NN * DH;                         // 6.4M
    const int embBlocks  = (int)((nodeElems + 255) / 256);

    // msg-kernel dynamic LDS: 2 z tiles + We/bee + ew/dst  (~25 KB)
    const size_t msgLds =
        (size_t)(2 * 16 * ZW) * sizeof(unsigned short) +
        (size_t)(DIN_E * DE + DE + 32) * sizeof(float) +
        (size_t)32 * sizeof(int);
    const int msgBlocks = 4096;

    // 1) node embedding -> hA (f32), hbf (bf16)
    k_embed_nodes<<<embBlocks, 256, 0, stream>>>(x, Wx, bx, hA, hbf);

    // 2) block 1: covalent
    k_zero<<<4096, 256, 0, stream>>>(agg, nodeElems);
    k_msg<<<msgBlocks, 256, msgLds, stream>>>(hbf, edge_attr, edge_idx,
                                              We, bee, Wf1, bf1, Ws1, bs1,
                                              agg, /*mode=*/0);
    k_post<<<NN, 128, 0, stream>>>(agg, hA, g1, beta1, Wl1, bl1, hB, hbf);

    // 3) block 2: non-covalent
    k_zero<<<4096, 256, 0, stream>>>(agg, nodeElems);
    k_msg<<<msgBlocks, 256, msgLds, stream>>>(hbf, edge_attr, edge_idx,
                                              We, bee, Wf2, bf2, Ws2, bs2,
                                              agg, /*mode=*/1);
    k_post<<<NN, 128, 0, stream>>>(agg, hB, g2, beta2, Wl2, bl2, hA, hbf);

    // 4) readout + pooling + energy
    k_zero<<<8, 256, 0, stream>>>(pooled, (long)BB * (DH / 4));
    k_readout<<<NN, 64, 0, stream>>>(hA, x, batch, Wm1, bm1, Wm2, bm2, pooled);
    k_energy<<<1, 64, 0, stream>>>(pooled, Wt, n_atom, out);
}
